// QuantumInspiredAttention_19576460935323
// MI455X (gfx1250) — matrix operand order repaired and checked by reading, not compiled, not run on hardware
//
#include <hip/hip_runtime.h>

typedef unsigned short u16;
typedef __attribute__((ext_vector_type(16))) __bf16 v16bf;
typedef __attribute__((ext_vector_type(8)))  float  v8f;

#define B_  2
#define S_  768
#define D_  1024
#define H_  16
#define HD_ 64
#define NQ_ 8
#define KQ_ 32    // quantum K (16 real + 16 zero pad -> one WMMA K-step)
#define VTR 128   // Vt padded rows per (b,h) so B-tile loads need no guards

// ---------------------------------------------------------------- helpers

__device__ __forceinline__ u16 f2bf(float x) {
  unsigned u = __float_as_uint(x);
  unsigned r = (u + 0x7FFFu + ((u >> 16) & 1u)) >> 16;   // round-to-nearest-even
  return (u16)r;
}
__device__ __forceinline__ float bf2f(u16 h) {
  return __uint_as_float(((unsigned)h) << 16);
}
__device__ __forceinline__ float sigmoidf_(float x) { return 1.0f / (1.0f + __expf(-x)); }

union Frag16 { unsigned int u[8]; v16bf v; };

__device__ __forceinline__ v8f wmma_bf16(const Frag16& a, const Frag16& b, v8f c) {
  return __builtin_amdgcn_wmma_f32_16x16x32_bf16(false, a.v, false, b.v, (short)0, c,
                                                 false, false);
}

// Generic pointer into __shared__ has the wave-relative LDS byte offset in its
// low 32 bits (flat LDS aperture, selected by the address's high 32 bits).
__device__ __forceinline__ unsigned lds_addr32(const void* p) {
  return (unsigned)(unsigned long long)p;
}

// CDNA5 async cache->LDS DMA (ASYNCcnt), GV mode: 64-bit VGPR address, no VGPR
// round-trip and no serializing loadcnt wait between global load and ds_store.
__device__ __forceinline__ void async_load_b128(unsigned ldsOff, const void* gaddr) {
  asm volatile("global_load_async_to_lds_b128 %0, %1, off"
               :: "v"(ldsOff), "v"(gaddr) : "memory");
}
__device__ __forceinline__ void wait_async0() {
  asm volatile("s_wait_asynccnt 0x0" ::: "memory");
}

// Issue async copies of a ROWSx32 bf16 tile (row-major along K) into LDS with
// row stride 40 ushorts (80B -> 16B-aligned b128 targets). 256 threads, no guards.
template<int ROWS>
__device__ __forceinline__ void load_tile_async(u16 (*dst)[40], const u16* __restrict__ src,
                                                int ld, int kk, int rowBaseG) {
  int tid = threadIdx.x;
#pragma unroll
  for (int i = 0; i < ROWS / 64; ++i) {
    int r = (tid >> 2) + i * 64;
    int c = (tid & 3) * 8;
    async_load_b128(lds_addr32(&dst[r][c]),
                    &src[(long long)(rowBaseG + r) * ld + kk + c]);
  }
}

// A fragment (16x32 bf16): lanes 0-15 -> M=lane, K dwords {0..3, 8..11};
// lanes 16-31 -> M=lane-16, K dwords {4..7, 12..15}. (ISA 7.12.2)
__device__ __forceinline__ void load_fragA(Frag16& f, const u16 (*As)[40],
                                           int rowBase, int l, int hi) {
  const u16* ar = &As[rowBase + l][0];
#pragma unroll
  for (int j = 0; j < 8; ++j) {
    int d = hi * 4 + (j & 3) + ((j & 4) ? 8 : 0);
    f.u[j] = *(const unsigned int*)(ar + 2 * d);
  }
}

// B fragment (32x16 bf16, stored as Bt[n][k]): lane = column N; lanes 0-15 hold
// K dwords {0..3, 8..11}, lanes 16-31 hold K dwords {4..7, 12..15} (matches A).
__device__ __forceinline__ void load_fragB(Frag16& f, const u16 (*Bs)[40],
                                           int colBase, int l, int hi) {
  const u16* br = &Bs[colBase + l][0];
#pragma unroll
  for (int j = 0; j < 8; ++j) {
    int d = hi * 4 + (j & 3) + ((j & 4) ? 8 : 0);
    f.u[j] = *(const unsigned int*)(br + 2 * d);
  }
}

__device__ __forceinline__ float block_reduce(float v, bool isMax, float* sm) {
#pragma unroll
  for (int o = 16; o; o >>= 1) {
    float t = __shfl_xor(v, o, 32);
    v = isMax ? fmaxf(v, t) : v + t;
  }
  int w = threadIdx.x >> 5;
  __syncthreads();                       // protect sm from previous use
  if ((threadIdx.x & 31) == 0) sm[w] = v;
  __syncthreads();
  float r = sm[0];
#pragma unroll
  for (int i = 1; i < 8; ++i) r = isMax ? fmaxf(r, sm[i]) : r + sm[i];
  return r;
}

// ---------------------------------------------------------------- conversion

__global__ __launch_bounds__(256) void cvt_bf16_kernel(const float* __restrict__ src,
                                                       u16* __restrict__ dst, int n) {
  int gid = blockIdx.x * 256 + threadIdx.x;
  if (gid < n) dst[gid] = f2bf(src[gid]);
}

// W[k][n] (KxN f32) -> Wt[n][k] bf16 so every GEMM is NT (contiguous-K B reads).
__global__ __launch_bounds__(256) void cvt_bf16_T_kernel(const float* __restrict__ src,
                                                         u16* __restrict__ dst,
                                                         int K, int N) {
  int gid = blockIdx.x * 256 + threadIdx.x;
  if (gid >= K * N) return;
  int n = gid % N, k = gid / N;
  dst[(long long)n * K + k] = f2bf(src[(long long)k * N + n]);
}

// Vbf [B,S,H*HD] -> Vt [B,H,VTR,S] bf16, rows HD..VTR-1 zero-filled so the
// attn@V B-tile (128 rows) loads are unguarded straight-line code.
__global__ __launch_bounds__(256) void transpose_v_kernel(const u16* __restrict__ src,
                                                          u16* __restrict__ dst) {
  int gid = blockIdx.x * 256 + threadIdx.x;   // enumerates (b,h,row,s), s fastest
  if (gid >= B_ * H_ * VTR * S_) return;
  int s = gid % S_; int r = gid / S_;
  int hd = r % VTR; r /= VTR;
  int h = r % H_;  int b = r / H_;
  u16 v = 0;
  if (hd < HD_) v = src[((long long)(b * S_ + s)) * D_ + h * HD_ + hd];
  dst[gid] = v;
}

// ---------------------------------------------------------------- generic NT GEMM
// C[m][n] = sum_k A[m][k] * Bt[n][k]  (+bias[n]); bf16 in, f32 accumulate.
// 256 threads = 8 waves (wave32), WG tile 64x128, wave tile 32x32 (2x2 wmma frags).
// Tiles staged via global_load_async_to_lds_b128; M and tile-N always in bounds.

__global__ __launch_bounds__(256) void gemm_nt_kernel(
    const u16* __restrict__ A, const u16* __restrict__ Bt,
    float* __restrict__ Cf, u16* __restrict__ Cbf, const float* __restrict__ bias,
    int M, int N, int K, int lda, int ldb, int ldc,
    long long sA, long long sB, int hpb, long long sCb, long long sCh) {
  __shared__ u16 As[64][40];
  __shared__ u16 Bs[128][40];
  int z = blockIdx.z;
  const u16* Ap = A + (long long)z * sA;
  const u16* Bp = Bt + (long long)z * sB;
  long long cbase = (long long)(z / hpb) * sCb + (long long)(z % hpb) * sCh;
  int mBase = blockIdx.y * 64, nBase = blockIdx.x * 128;
  int tid = threadIdx.x, lane = tid & 31, wave = tid >> 5;
  int wm = wave & 1, wn = wave >> 1;     // 2 waves along M, 4 along N
  int l = lane & 15, hi = lane >> 4;
  v8f acc[2][2];
#pragma unroll
  for (int i = 0; i < 2; ++i)
#pragma unroll
    for (int j = 0; j < 2; ++j) acc[i][j] = (v8f){0, 0, 0, 0, 0, 0, 0, 0};

  for (int kk = 0; kk < K; kk += 32) {
    load_tile_async<64>(As, Ap, lda, kk, mBase);
    load_tile_async<128>(Bs, Bp, ldb, kk, nBase);
    if (kk + 32 < K) {                    // global_prefetch_b8 of next K tile
      __builtin_prefetch(&Ap[(long long)(mBase + (tid >> 2)) * lda + kk + 32], 0, 0);
      __builtin_prefetch(&Bp[(long long)(nBase + (tid >> 1)) * ldb + kk + 32], 0, 0);
    }
    wait_async0();                        // this wave's DMAs done
    __syncthreads();                      // all waves' DMAs visible
    Frag16 fa[2], fb[2];
    load_fragA(fa[0], As, wm * 32, l, hi);
    load_fragA(fa[1], As, wm * 32 + 16, l, hi);
    load_fragB(fb[0], Bs, wn * 32, l, hi);
    load_fragB(fb[1], Bs, wn * 32 + 16, l, hi);
#pragma unroll
    for (int fm = 0; fm < 2; ++fm)
#pragma unroll
      for (int fn = 0; fn < 2; ++fn) acc[fm][fn] = wmma_bf16(fa[fm], fb[fn], acc[fm][fn]);
    __syncthreads();                      // LDS free before next iteration's DMAs
  }

#pragma unroll
  for (int fm = 0; fm < 2; ++fm)
#pragma unroll
    for (int fn = 0; fn < 2; ++fn)
#pragma unroll
      for (int r = 0; r < 8; ++r) {
        int gm = mBase + wm * 32 + fm * 16 + hi * 8 + r;   // C layout: M = hi*8+vgpr
        int gn = nBase + wn * 32 + fn * 16 + l;            //           N = lane&15
        if (gn < N) {                                      // only attn@V (N=64) trims
          float v = acc[fm][fn][r];
          if (bias) v += bias[gn];
          long long ci = cbase + (long long)gm * ldc + gn;
          if (Cf)  Cf[ci] = v;
          if (Cbf) Cbf[ci] = f2bf(v);
        }
      }
}

// ---------------------------------------------------------------- quantum factor map
// Qq = tanh(Q@Wm+b); A[i] = sig(w)/NQ * [cos(Qq+p), sin(Qq+p)]; B[j] = [cos(Kq), sin(Kq)]
// so  mean_n(cos(Qq_i - Kq_j + p_n) * sig(w_n)) == A_i . B_j   (rank-16, padded to 32)

__global__ __launch_bounds__(256) void qmap_kernel(
    const float* __restrict__ Qf, const float* __restrict__ Kf,
    const float* __restrict__ W, const float* __restrict__ bias,
    const float* __restrict__ qw, const float* __restrict__ ph,
    u16* __restrict__ Aq, u16* __restrict__ Bq) {
  int gid = blockIdx.x * 256 + threadIdx.x;   // gid = (b*H+h)*S + s
  if (gid >= B_ * H_ * S_) return;
  int s = gid % S_; int t = gid / S_;
  int h = t % H_;   int b = t / H_;
  const float* qrow = Qf + ((long long)(b * S_ + s)) * D_ + h * HD_;
  const float* krow = Kf + ((long long)(b * S_ + s)) * D_ + h * HD_;
  float qa[NQ_], ka[NQ_];
#pragma unroll
  for (int n = 0; n < NQ_; ++n) { qa[n] = bias[n]; ka[n] = bias[n]; }
  for (int d = 0; d < HD_; ++d) {
    float qv = qrow[d], kv = krow[d];
#pragma unroll
    for (int n = 0; n < NQ_; ++n) {
      float w = W[d * NQ_ + n];
      qa[n] = fmaf(qv, w, qa[n]);
      ka[n] = fmaf(kv, w, ka[n]);
    }
  }
  u16* arow = Aq + (long long)gid * KQ_;
  u16* brow = Bq + (long long)gid * KQ_;
#pragma unroll
  for (int n = 0; n < NQ_; ++n) {
    float qq = tanhf(qa[n]);
    float kq = tanhf(ka[n]);
    float w = sigmoidf_(qw[h * NQ_ + n]) * (1.0f / NQ_);
    float sA, cA, sB, cB;
    __sincosf(qq + ph[h * NQ_ + n], &sA, &cA);
    __sincosf(kq, &sB, &cB);
    arow[n] = f2bf(w * cA); arow[NQ_ + n] = f2bf(w * sA);
    brow[n] = f2bf(cB);     brow[NQ_ + n] = f2bf(sB);
  }
#pragma unroll
  for (int n = 2 * NQ_; n < KQ_; ++n) { arow[n] = 0; brow[n] = 0; }
}

// ---------------------------------------------------------------- fused score GEMM
// classical (K=64, from Q/K bf16) and quantum (K=32, from Aq/Bq) tiles share one
// epilogue applying (1-mix)/sqrt(HD), mix, and exp(-0.1|i-j|).

__global__ __launch_bounds__(256) void scores_kernel(
    const u16* __restrict__ Qbf, const u16* __restrict__ Kbf,
    const u16* __restrict__ Aq, const u16* __restrict__ Bq,
    const float* __restrict__ entp, u16* __restrict__ out) {
  __shared__ u16 As[64][40];
  __shared__ u16 Bs[128][40];
  int z = blockIdx.z; int b = z >> 4; int h = z & 15;
  const u16* Ac  = Qbf + (long long)b * S_ * D_ + h * HD_;
  const u16* Bc  = Kbf + (long long)b * S_ * D_ + h * HD_;
  const u16* Aqp = Aq + (long long)z * S_ * KQ_;
  const u16* Bqp = Bq + (long long)z * S_ * KQ_;
  int mBase = blockIdx.y * 64, nBase = blockIdx.x * 128;
  int tid = threadIdx.x, lane = tid & 31, wave = tid >> 5;
  int wm = wave & 1, wn = wave >> 1;
  int l = lane & 15, hi = lane >> 4;
  v8f accC[2][2], accQ[2][2];
#pragma unroll
  for (int i = 0; i < 2; ++i)
#pragma unroll
    for (int j = 0; j < 2; ++j) {
      accC[i][j] = (v8f){0, 0, 0, 0, 0, 0, 0, 0};
      accQ[i][j] = (v8f){0, 0, 0, 0, 0, 0, 0, 0};
    }

  Frag16 fa[2], fb[2];
#pragma unroll
  for (int kk = 0; kk < HD_; kk += 32) {        // classical: 2 K-steps
    load_tile_async<64>(As, Ac, D_, kk, mBase);
    load_tile_async<128>(Bs, Bc, D_, kk, nBase);
    wait_async0();
    __syncthreads();
    load_fragA(fa[0], As, wm * 32, l, hi);
    load_fragA(fa[1], As, wm * 32 + 16, l, hi);
    load_fragB(fb[0], Bs, wn * 32, l, hi);
    load_fragB(fb[1], Bs, wn * 32 + 16, l, hi);
#pragma unroll
    for (int fm = 0; fm < 2; ++fm)
#pragma unroll
      for (int fn = 0; fn < 2; ++fn) accC[fm][fn] = wmma_bf16(fa[fm], fb[fn], accC[fm][fn]);
    __syncthreads();
  }
  // quantum: single K=32 step
  load_tile_async<64>(As, Aqp, KQ_, 0, mBase);
  load_tile_async<128>(Bs, Bqp, KQ_, 0, nBase);
  wait_async0();
  __syncthreads();
  load_fragA(fa[0], As, wm * 32, l, hi);
  load_fragA(fa[1], As, wm * 32 + 16, l, hi);
  load_fragB(fb[0], Bs, wn * 32, l, hi);
  load_fragB(fb[1], Bs, wn * 32 + 16, l, hi);
#pragma unroll
  for (int fm = 0; fm < 2; ++fm)
#pragma unroll
    for (int fn = 0; fn < 2; ++fn) accQ[fm][fn] = wmma_bf16(fa[fm], fb[fn], accQ[fm][fn]);

  float mix = sigmoidf_(entp[0]);
  float c1 = (1.0f - mix) * 0.125f;             // (1-mix)/sqrt(64)
  u16* orow = out + (long long)z * S_ * S_;
#pragma unroll
  for (int fm = 0; fm < 2; ++fm)
#pragma unroll
    for (int fn = 0; fn < 2; ++fn)
#pragma unroll
      for (int r = 0; r < 8; ++r) {
        int gm = mBase + wm * 32 + fm * 16 + hi * 8 + r;
        int gn = nBase + wn * 32 + fn * 16 + l;
        float ent = __expf(-0.1f * fabsf((float)(gm - gn)));
        float sval = c1 * accC[fm][fn][r] + mix * ent * accQ[fm][fn][r];
        orow[(long long)gm * S_ + gn] = f2bf(sval);
      }
}

// ---------------------------------------------------------------- softmax + head-mean
// One block per (b, i): loops all 16 heads, normalizes rows in place (bf16),
// accumulates mean over heads into d_out's second region. No atomics.

__global__ __launch_bounds__(256) void softmax_mean_kernel(u16* __restrict__ attn,
                                                           float* __restrict__ meanOut) {
  __shared__ float sm[8];
  int bi = blockIdx.x;              // b*S + i
  int b = bi / S_, i = bi % S_;
  int tid = threadIdx.x;
  float meanAcc[3] = {0.f, 0.f, 0.f};
  for (int h = 0; h < H_; ++h) {
    u16* row = attn + (((long long)(b * H_ + h) * S_ + i) * S_);
    float v[3];
#pragma unroll
    for (int t = 0; t < 3; ++t) v[t] = bf2f(row[tid + t * 256]);
    float m = fmaxf(v[0], fmaxf(v[1], v[2]));
    m = block_reduce(m, true, sm);
    float e[3], ssum = 0.f;
#pragma unroll
    for (int t = 0; t < 3; ++t) { e[t] = __expf(v[t] - m); ssum += e[t]; }
    ssum = block_reduce(ssum, false, sm);
    float inv = 1.0f / ssum;
#pragma unroll
    for (int t = 0; t < 3; ++t) {
      float p = e[t] * inv;
      row[tid + t * 256] = f2bf(p);
      meanAcc[t] += p * (1.0f / H_);
    }
  }
  float* mrow = meanOut + (long long)bi * S_;
#pragma unroll
  for (int t = 0; t < 3; ++t) mrow[tid + t * 256] = meanAcc[t];
}

// ---------------------------------------------------------------- host launch

extern "C" void kernel_launch(void* const* d_in, const int* in_sizes, int n_in,
                              void* d_out, int out_size, void* d_ws, size_t ws_size,
                              hipStream_t stream) {
  (void)in_sizes; (void)n_in; (void)out_size; (void)ws_size;
  const float* query = (const float*)d_in[0];
  const float* key_  = (const float*)d_in[1];
  const float* value = (const float*)d_in[2];
  const float* Wq = (const float*)d_in[3];  const float* bq = (const float*)d_in[4];
  const float* Wk = (const float*)d_in[5];  const float* bk = (const float*)d_in[6];
  const float* Wv = (const float*)d_in[7];  const float* bv = (const float*)d_in[8];
  const float* Wo = (const float*)d_in[9];  const float* bo = (const float*)d_in[10];
  const float* qmapW = (const float*)d_in[11];
  const float* qmapB = (const float*)d_in[12];
  const float* qw    = (const float*)d_in[13];
  const float* ph    = (const float*)d_in[14];
  const float* entp  = (const float*)d_in[15];

  char* ws = (char*)d_ws;
  size_t off = 0;
  auto alloc = [&](size_t bytes) -> char* {
    char* p = ws + off;
    off = (off + bytes + 255) & ~(size_t)255;
    return p;
  };
  const size_t BSD = (size_t)B_ * S_ * D_;
  u16* wTq = (u16*)alloc((size_t)D_ * D_ * 2);
  u16* wTk = (u16*)alloc((size_t)D_ * D_ * 2);
  u16* wTv = (u16*)alloc((size_t)D_ * D_ * 2);
  u16* wTo = (u16*)alloc((size_t)D_ * D_ * 2);
  u16* qxb = (u16*)alloc(BSD * 2);             // bf16 inputs; reused as attended later
  u16* kxb = (u16*)alloc(BSD * 2);
  u16* vxb = (u16*)alloc(BSD * 2);
  float* Qf = (float*)alloc(BSD * 4);
  float* Kf = (float*)alloc(BSD * 4);
  u16* Qbf = (u16*)alloc(BSD * 2);
  u16* Kbf = (u16*)alloc(BSD * 2);
  u16* Vbf = (u16*)alloc(BSD * 2);
  u16* Vt  = (u16*)alloc((size_t)B_ * H_ * VTR * S_ * 2);   // zero-padded to 128 rows
  u16* Aq  = (u16*)alloc((size_t)B_ * H_ * S_ * KQ_ * 2);
  u16* Bq  = (u16*)alloc((size_t)B_ * H_ * S_ * KQ_ * 2);
  u16* attn = (u16*)alloc((size_t)B_ * H_ * S_ * S_ * 2);   // scores, then attn in place

  float* outp  = (float*)d_out;
  float* meanp = outp + BSD;

  // 1) weights -> bf16, transposed (NT GEMM layout)
  dim3 blk(256);
  dim3 gW((D_ * D_ + 255) / 256);
  cvt_bf16_T_kernel<<<gW, blk, 0, stream>>>(Wq, wTq, D_, D_);
  cvt_bf16_T_kernel<<<gW, blk, 0, stream>>>(Wk, wTk, D_, D_);
  cvt_bf16_T_kernel<<<gW, blk, 0, stream>>>(Wv, wTv, D_, D_);
  cvt_bf16_T_kernel<<<gW, blk, 0, stream>>>(Wo, wTo, D_, D_);

  // 2) inputs -> bf16
  dim3 gX((unsigned)((BSD + 255) / 256));
  cvt_bf16_kernel<<<gX, blk, 0, stream>>>(query, qxb, (int)BSD);
  cvt_bf16_kernel<<<gX, blk, 0, stream>>>(key_,  kxb, (int)BSD);
  cvt_bf16_kernel<<<gX, blk, 0, stream>>>(value, vxb, (int)BSD);

  // 3) projections: [1536,1024] @ Wt^T  -> f32 (for qmap) + bf16 (for WMMA)
  dim3 gP(D_ / 128, (B_ * S_) / 64, 1);
  gemm_nt_kernel<<<gP, blk, 0, stream>>>(qxb, wTq, Qf, Qbf, bq,
      B_ * S_, D_, D_, D_, D_, D_, 0, 0, 1, 0, 0);
  gemm_nt_kernel<<<gP, blk, 0, stream>>>(kxb, wTk, Kf, Kbf, bk,
      B_ * S_, D_, D_, D_, D_, D_, 0, 0, 1, 0, 0);
  gemm_nt_kernel<<<gP, blk, 0, stream>>>(vxb, wTv, nullptr, Vbf, bv,
      B_ * S_, D_, D_, D_, D_, D_, 0, 0, 1, 0, 0);

  // 4) V -> [B,H,128,S] (zero-padded) for unguarded NT attn@V
  dim3 gVT((unsigned)(((size_t)B_ * H_ * VTR * S_ + 255) / 256));
  transpose_v_kernel<<<gVT, blk, 0, stream>>>(Vbf, Vt);

  // 5) quantum factor map (rank-16 factorization of the cos-phase term)
  dim3 gQ((B_ * H_ * S_ + 255) / 256);
  qmap_kernel<<<gQ, blk, 0, stream>>>(Qf, Kf, qmapW, qmapB, qw, ph, Aq, Bq);

  // 6) fused scores: classical QK^T + quantum GEMM + decay/mix epilogue
  dim3 gS(S_ / 128, S_ / 64, B_ * H_);
  scores_kernel<<<gS, blk, 0, stream>>>(Qbf, Kbf, Aq, Bq, entp, attn);

  // 7) softmax rows + mean over heads (writes second output region)
  softmax_mean_kernel<<<dim3(B_ * S_), blk, 0, stream>>>(attn, meanp);

  // 8) attended = attn @ V  (per b,h; C scattered into [B,S,D] bf16, reuse qxb)
  u16* attended = qxb;
  dim3 gAV(1, S_ / 64, B_ * H_);
  gemm_nt_kernel<<<gAV, blk, 0, stream>>>(attn, Vt, nullptr, attended, nullptr,
      S_, HD_, S_, S_, S_, D_,
      (long long)S_ * S_, (long long)VTR * S_,
      H_, (long long)S_ * D_, (long long)HD_);

  // 9) out = attended @ Wo + bo  -> d_out (f32)
  gemm_nt_kernel<<<gP, blk, 0, stream>>>(attended, wTo, outp, nullptr, bo,
      B_ * S_, D_, D_, D_, D_, D_, 0, 0, 1, 0, 0);
}
